// tworing_conv_layer_22565758173779
// MI455X (gfx1250) — matrix operand order repaired and checked by reading, not compiled, not run on hardware
//
#include <hip/hip_runtime.h>

// ---------------------------------------------------------------------------
// Two-ring icosphere conv: y[N,64] = gather(x, neigh)[N,19*64] @ W^T + b
// GEMM M=163842, K=1216, N=64 on bf16 WMMA (f32 accumulate).
// B operand (swizzled W, 152 KB) staged once per block in CDNA5's 320KB LDS.
// ---------------------------------------------------------------------------

typedef __attribute__((ext_vector_type(16))) __bf16        v16bf;
typedef __attribute__((ext_vector_type(8)))  float         v8f;
typedef __attribute__((ext_vector_type(4)))  unsigned int  u32x4;

#define NV     163842
#define CIN    64
#define COUT   64
#define KN     19
#define KDIM   (KN * CIN)          // 1216
#define NCHUNK (KDIM / 32)         // 38 K-chunks of 32
#define WFRAG_ELEMS (NCHUNK * 4 * 32 * 16)   // 77824 bf16 = 155,648 bytes

// ---------------------------------------------------------------------------
// fp32 -> bf16 (round to nearest even), pure bit ops
// ---------------------------------------------------------------------------
__device__ __forceinline__ unsigned short f32_to_bf16(float f) {
    unsigned u = __float_as_uint(f);
    u += 0x7FFFu + ((u >> 16) & 1u);
    return (unsigned short)(u >> 16);
}

// ---------------------------------------------------------------------------
// Prep 1: x [N*64] fp32 -> bf16 (4 elements / thread)
// ---------------------------------------------------------------------------
__global__ void cvt_x_kernel(const float* __restrict__ x,
                             unsigned short* __restrict__ xbf, int n) {
    int i = (blockIdx.x * blockDim.x + threadIdx.x) * 4;
#pragma unroll
    for (int j = 0; j < 4; ++j)
        if (i + j < n) xbf[i + j] = f32_to_bf16(x[i + j]);
}

// ---------------------------------------------------------------------------
// Prep 2: swizzle W [64 x 1216] into per-chunk bf16 B-fragment order.
// wfrag[(((kk*4 + t)*32 + lane)*16) + e]:
//   lane n (n=lane&15, hi=lane>>4) of tile t holds, for column o = t*16+n,
//   K values kg = kk*32 + hi*16 + e (e=0..15) contiguously
//   (v16bf B operand of V_WMMA_F32_16X16X32_BF16).
// ---------------------------------------------------------------------------
__global__ void pack_w_kernel(const float* __restrict__ W,
                              unsigned short* __restrict__ wfrag) {
    int idx = blockIdx.x * 256 + threadIdx.x;      // 0 .. 77823
    int e    = idx & 15;
    int lane = (idx >> 4) & 31;
    int t    = (idx >> 9) & 3;
    int kk   = idx >> 11;
    int o    = t * 16 + (lane & 15);
    int kg   = kk * 32 + (lane >> 4) * 16 + e;
    wfrag[idx] = f32_to_bf16(W[o * KDIM + kg]);
}

// ---------------------------------------------------------------------------
// Main: each wave computes a 32-row M tile x 64 output columns.
// B fragments come from LDS; A fragments are bf16 gathers from global (L2).
// ---------------------------------------------------------------------------
__global__ __launch_bounds__(256)
void tworing_wmma_kernel(const int* __restrict__ neigh,
                         const float* __restrict__ bias,
                         const unsigned short* __restrict__ xbf_,
                         const unsigned short* __restrict__ wfrag_,
                         float* __restrict__ out) {
    const __bf16* __restrict__ xbf = (const __bf16*)xbf_;

    // ---- stage the whole swizzled W panel (152 KB) into LDS, once per block
    __shared__ unsigned short wlds[WFRAG_ELEMS];
    {
        const u32x4* __restrict__ src = (const u32x4*)wfrag_;
        u32x4* dst = (u32x4*)wlds;
        const int nvec = WFRAG_ELEMS / 8;          // 9728 x 16B
        for (int i = threadIdx.x; i < nvec; i += 256)
            dst[i] = src[i];
    }
    __syncthreads();

    const __bf16* wb = (const __bf16*)wlds;

    const int lane  = threadIdx.x & 31;
    const int wave  = threadIdx.x >> 5;
    const int tileM = blockIdx.x * 8 + wave;       // 32-row tile id
    const int row0  = tileM * 32;
    if (row0 >= NV) return;                        // wave-uniform exit

    const int m  = lane & 15;                      // A row / output col-in-tile
    const int hi = lane >> 4;                      // K half selector

    // gather rows for the two 16-row sub-tiles (clamped for the tail)
    const int rowA  = row0 + m;
    const int rowB  = row0 + 16 + m;
    const int rowAc = (rowA < NV) ? rowA : (NV - 1);
    const int rowBc = (rowB < NV) ? rowB : (NV - 1);

    v8f accA0 = {}, accA1 = {}, accA2 = {}, accA3 = {};
    v8f accB0 = {}, accB1 = {}, accB2 = {}, accB3 = {};

    int nb0 = neigh[rowAc * KN];                   // slot-0 neighbors
    int nb1 = neigh[rowBc * KN];
    for (int slot = 0; slot < KN; ++slot) {
        const int nxt = (slot + 1 < KN) ? slot + 1 : slot;
        const int nb0n = neigh[rowAc * KN + nxt];
        const int nb1n = neigh[rowBc * KN + nxt];
        __builtin_prefetch(xbf + (long)nb0n * CIN, 0, 0);
        __builtin_prefetch(xbf + (long)nb1n * CIN, 0, 0);

        const __bf16* xr0 = xbf + (long)nb0 * CIN;
        const __bf16* xr1 = xbf + (long)nb1 * CIN;
#pragma unroll
        for (int half = 0; half < 2; ++half) {     // two K=32 chunks / neighbor
            const int kk = slot * 2 + half;

            // ---- A fragments: lane holds K = hi*16 + 0..15 contiguously
            union { v16bf v; u32x4 q[2]; } a0, a1;
            const __bf16* ap0 = xr0 + half * 32 + hi * 16;
            const __bf16* ap1 = xr1 + half * 32 + hi * 16;
            a0.q[0] = *(const u32x4*)(ap0);
            a0.q[1] = *(const u32x4*)(ap0 + 8);
            a1.q[0] = *(const u32x4*)(ap1);
            a1.q[1] = *(const u32x4*)(ap1 + 8);

            // ---- B fragments from LDS: 32 contiguous bytes per lane
            const __bf16* wp = wb + ((kk * 4) * 32 + lane) * 16;
            union { v16bf v; u32x4 q[2]; } b0, b1, b2, b3;
            b0.q[0] = *(const u32x4*)(wp + 0 * 512);
            b0.q[1] = *(const u32x4*)(wp + 0 * 512 + 8);
            b1.q[0] = *(const u32x4*)(wp + 1 * 512);
            b1.q[1] = *(const u32x4*)(wp + 1 * 512 + 8);
            b2.q[0] = *(const u32x4*)(wp + 2 * 512);
            b2.q[1] = *(const u32x4*)(wp + 2 * 512 + 8);
            b3.q[0] = *(const u32x4*)(wp + 3 * 512);
            b3.q[1] = *(const u32x4*)(wp + 3 * 512 + 8);

            // ---- 8x v_wmma_f32_16x16x32_bf16, B reused across both A tiles
            accA0 = __builtin_amdgcn_wmma_f32_16x16x32_bf16(
                false, a0.v, false, b0.v, (short)0, accA0, false, false);
            accB0 = __builtin_amdgcn_wmma_f32_16x16x32_bf16(
                false, a1.v, false, b0.v, (short)0, accB0, false, false);
            accA1 = __builtin_amdgcn_wmma_f32_16x16x32_bf16(
                false, a0.v, false, b1.v, (short)0, accA1, false, false);
            accB1 = __builtin_amdgcn_wmma_f32_16x16x32_bf16(
                false, a1.v, false, b1.v, (short)0, accB1, false, false);
            accA2 = __builtin_amdgcn_wmma_f32_16x16x32_bf16(
                false, a0.v, false, b2.v, (short)0, accA2, false, false);
            accB2 = __builtin_amdgcn_wmma_f32_16x16x32_bf16(
                false, a1.v, false, b2.v, (short)0, accB2, false, false);
            accA3 = __builtin_amdgcn_wmma_f32_16x16x32_bf16(
                false, a0.v, false, b3.v, (short)0, accA3, false, false);
            accB3 = __builtin_amdgcn_wmma_f32_16x16x32_bf16(
                false, a1.v, false, b3.v, (short)0, accB3, false, false);
        }
        nb0 = nb0n;
        nb1 = nb1n;
    }

    // ---- epilogue: C/D layout lane l, reg r -> (M = r + 8*hi, N = l&15)
    v8f accs[2][4] = {{accA0, accA1, accA2, accA3},
                      {accB0, accB1, accB2, accB3}};
#pragma unroll
    for (int s = 0; s < 2; ++s) {
#pragma unroll
        for (int t = 0; t < 4; ++t) {
            const int col = t * 16 + m;
            const float bc = bias[col];
#pragma unroll
            for (int r = 0; r < 8; ++r) {
                const int orow = row0 + s * 16 + hi * 8 + r;
                if (orow < NV)
                    out[(long)orow * COUT + col] = accs[s][t][r] + bc;
            }
        }
    }
}

// ---------------------------------------------------------------------------
// Host-side launch
// ---------------------------------------------------------------------------
extern "C" void kernel_launch(void* const* d_in, const int* in_sizes, int n_in,
                              void* d_out, int out_size, void* d_ws, size_t ws_size,
                              hipStream_t stream) {
    const float* x     = (const float*)d_in[0];
    const int*   neigh = (const int*)d_in[1];
    const float* W     = (const float*)d_in[2];
    const float* bias  = (const float*)d_in[3];
    float*       out   = (float*)d_out;

    // workspace: [ x_bf16 : N*64 ushort (20.97 MB) ][ wfrag : 77824 ushort ]
    unsigned short* xbf   = (unsigned short*)d_ws;
    unsigned short* wfrag = xbf + (size_t)NV * CIN;

    // prep 1: convert x to bf16
    {
        const int n = NV * CIN;                    // 10,485,888
        const int threads = (n + 3) / 4;
        const int blocks  = (threads + 255) / 256;
        cvt_x_kernel<<<blocks, 256, 0, stream>>>(x, xbf, n);
    }
    // prep 2: swizzle W into fragment order (77824 elems = 304 blocks exactly)
    {
        pack_w_kernel<<<WFRAG_ELEMS / 256, 256, 0, stream>>>(W, wfrag);
    }
    // main GEMM: 32-row tiles, 8 waves/block -> 256 rows/block
    {
        const int tiles  = (NV + 31) / 32;         // 5121
        const int blocks = (tiles + 7) / 8;        // 641
        tworing_wmma_kernel<<<blocks, 256, 0, stream>>>(neigh, bias, xbf, wfrag, out);
    }
}
